// Decoder_52097953300775
// MI455X (gfx1250) — matrix operand order repaired and checked by reading, not compiled, not run on hardware
//
#include <hip/hip_runtime.h>
#include <cstdint>
#include <cstddef>

// ---------------------------------------------------------------------------
// Problem dimensions (fixed by the reference)
// ---------------------------------------------------------------------------
#define N_B   32      // batch
#define T_ENC 1000    // encoder length
#define L_DEC 200     // decoder length
#define VOCAB 4096
#define HDIM  512     // hidden
#define KSZ   512     // key size (== attention query size)
#define VSZ   256     // value size
#define GATES 2048    // 4*H (also 4*KS, identical here)
#define XCAT  768     // H + VS  (lstm1 input) and KS + VS (output-proj input)

typedef __bf16 bf16_t;
typedef __attribute__((ext_vector_type(8)))  __bf16 v8bf;
typedef __attribute__((ext_vector_type(16))) __bf16 v16bf;
typedef __attribute__((ext_vector_type(8)))  float  v8f;

__device__ __forceinline__ bf16_t f2bf(float f) { return (bf16_t)f; }
__device__ __forceinline__ float  sigf(float x) { return 1.0f / (1.0f + __expf(-x)); }

// ---------------------------------------------------------------------------
// Flat conversions / zero fill (one-time per launch)
// ---------------------------------------------------------------------------
__global__ void __launch_bounds__(256) cvt_bf16_kernel(const float* __restrict__ in,
                                                       bf16_t* __restrict__ out, int n) {
  int i = blockIdx.x * 256 + threadIdx.x;
  if (i < n) out[i] = f2bf(in[i]);
}
__global__ void __launch_bounds__(256) zero_f32_kernel(float* p, int n) {
  int i = blockIdx.x * 256 + threadIdx.x;
  if (i < n) p[i] = 0.0f;
}
__global__ void __launch_bounds__(256) zero_bf16_kernel(bf16_t* p, int n) {
  int i = blockIdx.x * 256 + threadIdx.x;
  if (i < n) p[i] = f2bf(0.0f);
}

// ---------------------------------------------------------------------------
// WMMA tile helpers (wave32; this code's own contiguous-k order, used for A and B alike):
//   A (16x32 bf16): lane 0..15 = row M with K 0..15 contiguous;
//                   lanes 16..31 same rows, K 16..31 (matches B).
//   B (32x16 bf16): lane = column (lane&15); lanes 0..15 hold K 0..15
//                   contiguous, lanes 16..31 hold K 16..31 -> one 32B load.
//   C/D (16x16 f32): lane 0..15 -> N=lane, vgpr r -> M=r; lanes 16..31 -> M=r+8.
// ---------------------------------------------------------------------------
__device__ __forceinline__ v16bf load_a_tile(const bf16_t* __restrict__ A, int lda,
                                             int row, int kt, int laneHi) {
  const bf16_t* p = A + (size_t)row * lda + kt;
  v8bf lo = *(const v8bf*)(p + (laneHi ? 16 : 0));
  v8bf hi = *(const v8bf*)(p + (laneHi ? 24 : 8));
  return __builtin_shufflevector(lo, hi, 0, 1, 2, 3, 4, 5, 6, 7,
                                         8, 9, 10, 11, 12, 13, 14, 15);
}
__device__ __forceinline__ v8f wmma_bf16(v16bf a, v16bf b, v8f c) {
  return __builtin_amdgcn_wmma_f32_16x16x32_bf16(false, a, false, b, (short)0, c,
                                                 false, false);
}

// ---------------------------------------------------------------------------
// Fused LSTM layer: gates GEMM (both input and recurrent matmuls) + cell.
//   gates(32 x 2048) = A1(32 x K1) @ W1^T + A2(32 x K2) @ W2^T + b1 + b2
// Each wave owns ONE 16-wide j-column tile but ALL FOUR gate quadrants
// (i,f,g,o at j, j+512, j+1024, j+1536) so the cell nonlinearity can be
// applied entirely in registers.  grid = 4 blocks x 256 (32 waves = 512 j).
// h is written as bf16 to up to two buffers (next GEMM inputs) + f32 query.
// NOTE: h output buffers must differ from A2 (caller double-buffers h).
// ---------------------------------------------------------------------------
__global__ void __launch_bounds__(256) lstm_fused_kernel(
    const bf16_t* __restrict__ A1, int lda1, const bf16_t* __restrict__ W1, int K1,
    const bf16_t* __restrict__ A2, int lda2, const bf16_t* __restrict__ W2, int K2,
    const float* __restrict__ bias1, const float* __restrict__ bias2,
    float* __restrict__ c,
    bf16_t* __restrict__ h_out1, int ld1,
    bf16_t* __restrict__ h_out2, int ld2,
    float* __restrict__ h_outF) {
  const int lane   = threadIdx.x & 31;
  const int wv     = threadIdx.x >> 5;
  const int laneHi = lane >> 4;
  const int lc     = lane & 15;
  const int jcol   = (blockIdx.x * 8 + wv) * 16 + lc;   // 0..511

  v8f acc[4][2];
#pragma unroll
  for (int q = 0; q < 4; ++q)
#pragma unroll
    for (int mt = 0; mt < 2; ++mt) acc[q][mt] = (v8f){0.f,0.f,0.f,0.f,0.f,0.f,0.f,0.f};

  for (int kt = 0; kt < K1; kt += 32) {
    v16bf a0 = load_a_tile(A1, lda1, lc,      kt, laneHi);
    v16bf a1 = load_a_tile(A1, lda1, lc + 16, kt, laneHi);
#pragma unroll
    for (int q = 0; q < 4; ++q) {
      v16bf b = *(const v16bf*)(W1 + (size_t)(q * 512 + jcol) * K1 + kt + (laneHi ? 16 : 0));
      acc[q][0] = wmma_bf16(a0, b, acc[q][0]);
      acc[q][1] = wmma_bf16(a1, b, acc[q][1]);
    }
  }
  for (int kt = 0; kt < K2; kt += 32) {
    v16bf a0 = load_a_tile(A2, lda2, lc,      kt, laneHi);
    v16bf a1 = load_a_tile(A2, lda2, lc + 16, kt, laneHi);
#pragma unroll
    for (int q = 0; q < 4; ++q) {
      v16bf b = *(const v16bf*)(W2 + (size_t)(q * 512 + jcol) * K2 + kt + (laneHi ? 16 : 0));
      acc[q][0] = wmma_bf16(a0, b, acc[q][0]);
      acc[q][1] = wmma_bf16(a1, b, acc[q][1]);
    }
  }

  float bi[4];
#pragma unroll
  for (int q = 0; q < 4; ++q)
    bi[q] = bias1[q * 512 + jcol] + bias2[q * 512 + jcol];

#pragma unroll
  for (int mt = 0; mt < 2; ++mt) {
#pragma unroll
    for (int r = 0; r < 8; ++r) {
      const int m = mt * 16 + laneHi * 8 + r;
      float gi = sigf (acc[0][mt][r] + bi[0]);
      float gf = sigf (acc[1][mt][r] + bi[1]);
      float gg = tanhf(acc[2][mt][r] + bi[2]);
      float go = sigf (acc[3][mt][r] + bi[3]);
      const size_t ci = (size_t)m * HDIM + jcol;
      float cn = gf * c[ci] + gi * gg;
      c[ci] = cn;
      float h = go * tanhf(cn);
      h_out1[(size_t)m * ld1 + jcol] = f2bf(h);
      if (h_out2) h_out2[(size_t)m * ld2 + jcol] = f2bf(h);
      if (h_outF) h_outF[ci] = h;
    }
  }
}

// ---------------------------------------------------------------------------
// Merged "attention(l) + output-projection(l-1)" kernel, 64 blocks x 256.
// proj(l-1) is off the recurrent dependency chain, so it overlaps with the
// attention of the next step inside one launch.
//   blocks  0..31: attention for step l (skipped when l == L_DEC)
//   blocks 32..63: projection for step l-1 (skipped when l == 0)
// Templated on key/value element type (bf16 staged copies, or f32 fallback
// if the workspace is too small to hold them).
// ---------------------------------------------------------------------------
template <typename KT>
__global__ void __launch_bounds__(256) attn_proj_kernel(
    const KT* __restrict__ key, const KT* __restrict__ value,
    const float* __restrict__ emb, const int* __restrict__ src_lens,
    const int* __restrict__ text, const float* __restrict__ h2F,
    bf16_t* __restrict__ xcat1, bf16_t* __restrict__ outin_cur,
    const bf16_t* __restrict__ outin_prev,
    const bf16_t* __restrict__ woutB, const float* __restrict__ b_out,
    float* __restrict__ out, int l) {
  const int tid = threadIdx.x;

  if (blockIdx.x < N_B) {
    // ---------------- attention for step l ----------------
    if (l >= L_DEC) return;
    __shared__ float sE[T_ENC];
    __shared__ float sH[KSZ];
    __shared__ float sred[256];
    const int n = blockIdx.x;

    for (int k = tid; k < KSZ; k += 256) sH[k] = h2F[n * KSZ + k];
    __syncthreads();

    const int wv = tid >> 5, lane = tid & 31;
    for (int t = wv; t < T_ENC; t += 8) {
      const KT* kp = key + ((size_t)n * T_ENC + t) * KSZ;
      float s = 0.0f;
      for (int k = lane; k < KSZ; k += 32) s += (float)kp[k] * sH[k];
      for (int off = 16; off > 0; off >>= 1) s += __shfl_xor(s, off, 32);
      if (lane == 0) sE[t] = s;
    }
    __syncthreads();

    const int slen = src_lens[n];
    float lm = -3.0e38f;
    for (int t = tid; t < T_ENC; t += 256) {
      float e = (t >= slen) ? -1.0e9f : sE[t];
      sE[t] = e;
      lm = fmaxf(lm, e);
    }
    sred[tid] = lm;
    __syncthreads();
    for (int s = 128; s > 0; s >>= 1) {
      if (tid < s) sred[tid] = fmaxf(sred[tid], sred[tid + s]);
      __syncthreads();
    }
    float mx = sred[0];
    __syncthreads();
    float ls = 0.0f;
    for (int t = tid; t < T_ENC; t += 256) {
      float e = __expf(sE[t] - mx);
      sE[t] = e;
      ls += e;
    }
    sred[tid] = ls;
    __syncthreads();
    for (int s = 128; s > 0; s >>= 1) {
      if (tid < s) sred[tid] += sred[tid + s];
      __syncthreads();
    }
    float inv = 1.0f / sred[0];
    __syncthreads();
    for (int t = tid; t < T_ENC; t += 256) sE[t] *= inv;
    __syncthreads();

    // context: thread tid owns value column tid (0..255)
    {
      float acc = 0.0f;
      const KT* vp = value + (size_t)n * T_ENC * VSZ + tid;
      for (int t = 0; t < T_ENC; ++t) acc += sE[t] * (float)vp[(size_t)t * VSZ];
      bf16_t cb = f2bf(acc);
      xcat1[n * XCAT + HDIM + tid]    = cb;
      outin_cur[n * XCAT + KSZ + tid] = cb;
    }

    // teacher-forced embedding gather
    const int tok = text[n * L_DEC + l];
    for (int h = tid; h < HDIM; h += 256)
      xcat1[n * XCAT + h] = f2bf(emb[(size_t)tok * HDIM + h]);
  } else {
    // ------------- output projection for step l-1 -------------
    if (l == 0) return;
    const int lane   = tid & 31;
    const int wv     = tid >> 5;
    const int laneHi = lane >> 4;
    const int lc     = lane & 15;
    const int ncol   = ((blockIdx.x - N_B) * 8 + wv) * 16 + lc;  // 0..4095

    v8f acc0 = (v8f){0.f,0.f,0.f,0.f,0.f,0.f,0.f,0.f};
    v8f acc1 = acc0;
    for (int kt = 0; kt < XCAT; kt += 32) {
      v16bf b  = *(const v16bf*)(woutB + (size_t)ncol * XCAT + kt + (laneHi ? 16 : 0));
      v16bf a0 = load_a_tile(outin_prev, XCAT, lc,      kt, laneHi);
      v16bf a1 = load_a_tile(outin_prev, XCAT, lc + 16, kt, laneHi);
      acc0 = wmma_bf16(a0, b, acc0);
      acc1 = wmma_bf16(a1, b, acc1);
    }
    const float bias = b_out[ncol];
    float* D = out + (size_t)(l - 1) * VOCAB;
    const long long ldd = (long long)L_DEC * VOCAB;
#pragma unroll
    for (int r = 0; r < 8; ++r) {
      D[(long long)(r + laneHi * 8) * ldd + ncol]      = acc0[r] + bias;
      D[(long long)(16 + r + laneHi * 8) * ldd + ncol] = acc1[r] + bias;
    }
  }
}

// ---------------------------------------------------------------------------
// Host-side orchestration
// ---------------------------------------------------------------------------
extern "C" void kernel_launch(void* const* d_in, const int* in_sizes, int n_in,
                              void* d_out, int out_size, void* d_ws, size_t ws_size,
                              hipStream_t stream) {
  const float* key    = (const float*)d_in[0];
  const float* value  = (const float*)d_in[1];
  const float* emb    = (const float*)d_in[2];
  const float* w_ih1  = (const float*)d_in[3];
  const float* w_hh1  = (const float*)d_in[4];
  const float* b_ih1  = (const float*)d_in[5];
  const float* b_hh1  = (const float*)d_in[6];
  const float* w_ih2  = (const float*)d_in[7];
  const float* w_hh2  = (const float*)d_in[8];
  const float* b_ih2  = (const float*)d_in[9];
  const float* b_hh2  = (const float*)d_in[10];
  const float* w_out  = (const float*)d_in[11];
  const float* b_out  = (const float*)d_in[12];
  const int* src_lens = (const int*)d_in[13];
  const int* text     = (const int*)d_in[14];
  float* out = (float*)d_out;
  (void)in_sizes; (void)n_in; (void)out_size;

  // workspace carve-up (256B aligned slices)
  char* ws = (char*)d_ws;
  size_t off = 0;
  auto carve = [&](size_t bytes) -> char* {
    char* p = ws + off;
    off = (off + bytes + 255) & ~(size_t)255;
    return p;
  };
  bf16_t* wih1B = (bf16_t*)carve((size_t)GATES * XCAT * 2);
  bf16_t* whh1B = (bf16_t*)carve((size_t)GATES * HDIM * 2);
  bf16_t* wih2B = (bf16_t*)carve((size_t)GATES * HDIM * 2);
  bf16_t* whh2B = (bf16_t*)carve((size_t)GATES * KSZ  * 2);
  bf16_t* woutB = (bf16_t*)carve((size_t)VOCAB * XCAT * 2);
  bf16_t* xcat1 = (bf16_t*)carve((size_t)N_B * XCAT * 2);
  bf16_t* outin0 = (bf16_t*)carve((size_t)N_B * XCAT * 2);   // [h2|ctx] parity 0
  bf16_t* outin1 = (bf16_t*)carve((size_t)N_B * XCAT * 2);   // [h2|ctx] parity 1
  bf16_t* h1B0  = (bf16_t*)carve((size_t)N_B * HDIM * 2);
  bf16_t* h1B1  = (bf16_t*)carve((size_t)N_B * HDIM * 2);
  bf16_t* h2B0  = (bf16_t*)carve((size_t)N_B * KSZ  * 2);
  bf16_t* h2B1  = (bf16_t*)carve((size_t)N_B * KSZ  * 2);
  float*  h2F   = (float*) carve((size_t)N_B * KSZ  * 4);
  float*  c1    = (float*) carve((size_t)N_B * HDIM * 4);
  float*  c2    = (float*) carve((size_t)N_B * KSZ  * 4);
  const size_t base_bytes = off;

  // bf16 staging of key/value (biggest per-step L2 traffic) if ws permits
  const size_t keyN = (size_t)N_B * T_ENC * KSZ;   // 16.384M
  const size_t valN = (size_t)N_B * T_ENC * VSZ;   //  8.192M
  const bool stageKV = (ws_size >= base_bytes + (keyN + valN) * 2 + 512);
  bf16_t* keyB = nullptr; bf16_t* valB = nullptr;
  if (stageKV) {
    keyB = (bf16_t*)carve(keyN * 2);
    valB = (bf16_t*)carve(valN * 2);
  }

  bf16_t* outinBuf[2] = {outin0, outin1};
  bf16_t* h1Buf[2]    = {h1B0, h1B1};
  bf16_t* h2Buf[2]    = {h2B0, h2B1};

  auto cvt = [&](const float* src, bf16_t* dst, size_t n) {
    cvt_bf16_kernel<<<(unsigned)((n + 255) / 256), 256, 0, stream>>>(src, dst, (int)n);
  };
  cvt(w_ih1, wih1B, (size_t)GATES * XCAT);
  cvt(w_hh1, whh1B, (size_t)GATES * HDIM);
  cvt(w_ih2, wih2B, (size_t)GATES * HDIM);
  cvt(w_hh2, whh2B, (size_t)GATES * KSZ);
  cvt(w_out, woutB, (size_t)VOCAB * XCAT);
  if (stageKV) { cvt(key, keyB, keyN); cvt(value, valB, valN); }

  zero_f32_kernel<<<(N_B * KSZ + 255) / 256, 256, 0, stream>>>(h2F, N_B * KSZ);
  zero_f32_kernel<<<(N_B * HDIM + 255) / 256, 256, 0, stream>>>(c1, N_B * HDIM);
  zero_f32_kernel<<<(N_B * KSZ + 255) / 256, 256, 0, stream>>>(c2, N_B * KSZ);
  zero_bf16_kernel<<<(N_B * HDIM + 255) / 256, 256, 0, stream>>>(h1B0, N_B * HDIM);
  zero_bf16_kernel<<<(N_B * HDIM + 255) / 256, 256, 0, stream>>>(h1B1, N_B * HDIM);
  zero_bf16_kernel<<<(N_B * KSZ + 255) / 256, 256, 0, stream>>>(h2B0, N_B * KSZ);
  zero_bf16_kernel<<<(N_B * KSZ + 255) / 256, 256, 0, stream>>>(h2B1, N_B * KSZ);

  for (int l = 0; l <= L_DEC; ++l) {
    const int cur = l & 1, prev = (l + 1) & 1;
    // attention(l) [blocks 0..31]  +  projection(l-1) [blocks 32..63]
    if (stageKV) {
      attn_proj_kernel<bf16_t><<<2 * N_B, 256, 0, stream>>>(
          keyB, valB, emb, src_lens, text, h2F, xcat1,
          outinBuf[cur], outinBuf[prev], woutB, b_out, out, l);
    } else {
      attn_proj_kernel<float><<<2 * N_B, 256, 0, stream>>>(
          key, value, emb, src_lens, text, h2F, xcat1,
          outinBuf[cur], outinBuf[prev], woutB, b_out, out, l);
    }
    if (l == L_DEC) break;  // final launch only flushes projection(199)

    // LSTM1: gates + cell fused; reads h1(prev), writes h1(cur)
    lstm_fused_kernel<<<4, 256, 0, stream>>>(
        xcat1, XCAT, wih1B, XCAT, h1Buf[prev], HDIM, whh1B, HDIM,
        b_ih1, b_hh1, c1, h1Buf[cur], HDIM, nullptr, 0, nullptr);
    // LSTM2: gates + cell fused; reads h2(prev), writes h2(cur) to
    // (a) next-step GEMM input, (b) proj input [h2|ctx], (c) f32 attn query
    lstm_fused_kernel<<<4, 256, 0, stream>>>(
        h1Buf[cur], HDIM, wih2B, HDIM, h2Buf[prev], KSZ, whh2B, KSZ,
        b_ih2, b_hh2, c2, h2Buf[cur], KSZ, outinBuf[cur], XCAT, h2F);
  }
}
